// VisualSeekerAdapter_76991583748286
// MI455X (gfx1250) — matrix operand order repaired and checked by reading, not compiled, not run on hardware
//
#include <hip/hip_runtime.h>
#include <cmath>

typedef __attribute__((ext_vector_type(16))) _Float16 v16h;
typedef __attribute__((ext_vector_type(8)))  float    v8f;
typedef __attribute__((ext_vector_type(4)))  unsigned int u32x4;
typedef __attribute__((ext_vector_type(8)))  int      i32x8;
typedef __attribute__((ext_vector_type(4)))  int      i32x4;

#define BB   16
#define NN   4096
#define CIN  1024
#define DD   64
#define MM   16
#define KSEL 64
#define ROWS (BB*NN)   // 65536

#if defined(__has_builtin)
#if __has_builtin(__builtin_amdgcn_tensor_load_to_lds) && __has_builtin(__builtin_amdgcn_s_wait_tensorcnt)
#define USE_TDM 1
#endif
#endif

__device__ __forceinline__ float gelu_tanh(float x) {
  float x3 = x * x * x;
  return 0.5f * x * (1.0f + tanhf(0.7978845608028654f * (x + 0.044715f * x3)));
}

// ---------------------------------------------------------------------------
// Kernel 0: pack W_down (1024x64) and W_up (64x1024) fp32 -> f16 B-fragments.
// Fragment (32K x 16N, f16): within-frag offset = lane*16 + h,
//   k = kt*32 + (h&7) + (h>=8 ? 16 : 0) + (lane>=16 ? 8 : 0), n = nt*16 + (lane & 15).
// ---------------------------------------------------------------------------
__global__ __launch_bounds__(256) void pack_weights_kernel(
    const float* __restrict__ Wd, const float* __restrict__ Wu,
    _Float16* __restrict__ fWd, _Float16* __restrict__ fWu) {
  int t = blockIdx.x * 256 + threadIdx.x;
  if (t < 65536) {                       // W_down: 32 kt x 4 nt fragments
    int f = t >> 9, w = t & 511;
    int lane = w >> 4, h = w & 15;
    int kt = f >> 2, nt = f & 3;
    int k = kt * 32 + (h & 7) + ((h & 8) ? 16 : 0) + ((lane >= 16) ? 8 : 0);
    int n = nt * 16 + (lane & 15);
    fWd[t] = (_Float16)Wd[k * DD + n];
  } else {                               // W_up: 2 kt x 64 nt fragments
    int t2 = t - 65536;
    int f = t2 >> 9, w = t2 & 511;
    int lane = w >> 4, h = w & 15;
    int kt = f >> 6, nt = f & 63;
    int k = kt * 32 + (h & 7) + ((h & 8) ? 16 : 0) + ((lane >= 16) ? 8 : 0);
    int n = nt * 16 + (lane & 15);
    fWu[t2] = (_Float16)Wu[k * CIN + n];
  }
}

// ---------------------------------------------------------------------------
// Kernel 1: act = gelu(X @ W_down + b_down), WMMA f16->f32.
// 512 blocks x 256 thr (8 waves). Block stages all W_down frags (128KB) in LDS
// via the Tensor Data Mover (single DMA, TENSORcnt-tracked); each wave then
// computes a 16-row x 64-col tile: 32 K-steps x 4 v_wmma_f32_16x16x32_f16.
// ---------------------------------------------------------------------------
__global__ __launch_bounds__(256) void down_gelu_kernel(
    const float* __restrict__ X, const float* __restrict__ bdown,
    const _Float16* __restrict__ fWd, float* __restrict__ act) {
  extern __shared__ _Float16 sW[];  // 65536 halves = 131072 bytes
  int t = threadIdx.x;

#if defined(USE_TDM)
  if (t < 32) {
    // TDM D# descriptor (ISA 8.3/8.4): 1-row tile of 32768 dwords (128KB),
    // data_size=4B, no padding/iterate/gather, groups 2-3 unused (tile_dim2=0).
    unsigned      ldsoff = (unsigned)(unsigned long long)(uintptr_t)(void*)sW; // addr[31:0] = LDS offset
    unsigned long long ga = (unsigned long long)(uintptr_t)fWd;
    u32x4 g0;
    g0[0] = 1u;                                              // count=1 (valid user D#)
    g0[1] = ldsoff;                                          // lds_addr
    g0[2] = (unsigned)(ga & 0xffffffffu);                    // global_addr[31:0]
    g0[3] = (unsigned)((ga >> 32) & 0x1ffffffu) | (2u << 30);// global_addr[56:32] | type=2
    i32x8 g1;
    g1[0] = (int)(2u << 16);        // workgroup_mask=0 | data_size=2 (4B)
    g1[1] = (int)(32768u << 16);    // atomic_barrier=0 | tensor_dim0 lo16 (32768)
    g1[2] = (int)(1u << 16);        // tensor_dim0 hi=0 | tensor_dim1 lo16 = 1
    g1[3] = (int)(32768u << 16);    // tensor_dim1 hi=0 | tile_dim0 = 32768
    g1[4] = 1;                      // tile_dim1=1 | tile_dim2=0 (unused)
    g1[5] = 32768;                  // tensor_dim0_stride lo32
    g1[6] = 0;                      // stride hi | tensor_dim1_stride lo
    g1[7] = 0;                      // tensor_dim1_stride hi
    i32x4 gz4 = {0, 0, 0, 0};
    i32x8 gz8 = {0, 0, 0, 0, 0, 0, 0, 0};
    __builtin_amdgcn_tensor_load_to_lds(g0, g1, gz4, gz4, gz8, 0);
    __builtin_amdgcn_s_wait_tensorcnt(0);
  }
  __syncthreads();
#else
  {
    const float4* src = reinterpret_cast<const float4*>(fWd);
    float4* dst = reinterpret_cast<float4*>(sW);
    for (int i = t; i < 8192; i += 256) dst[i] = src[i];
  }
  __syncthreads();
#endif

  int wave = t >> 5, lane = t & 31;
  int r0 = (blockIdx.x * 8 + wave) * 16;
  const float* xrow = X + (size_t)(r0 + (lane & 15)) * CIN;
  int kofs = (lane >= 16) ? 8 : 0;   // A-fragment K striping per half-wave

  v8f acc[4] = {};
  for (int kt = 0; kt < 32; ++kt) {
    int kb = kt * 32 + kofs;
    float4 fa = *reinterpret_cast<const float4*>(xrow + kb);
    float4 fb = *reinterpret_cast<const float4*>(xrow + kb + 4);
    float4 fc = *reinterpret_cast<const float4*>(xrow + kb + 16);
    float4 fd = *reinterpret_cast<const float4*>(xrow + kb + 20);
    v16h a;
    a[0]  = (_Float16)fa.x; a[1]  = (_Float16)fa.y; a[2]  = (_Float16)fa.z; a[3]  = (_Float16)fa.w;
    a[4]  = (_Float16)fb.x; a[5]  = (_Float16)fb.y; a[6]  = (_Float16)fb.z; a[7]  = (_Float16)fb.w;
    a[8]  = (_Float16)fc.x; a[9]  = (_Float16)fc.y; a[10] = (_Float16)fc.z; a[11] = (_Float16)fc.w;
    a[12] = (_Float16)fd.x; a[13] = (_Float16)fd.y; a[14] = (_Float16)fd.z; a[15] = (_Float16)fd.w;
#pragma unroll
    for (int j = 0; j < 4; ++j) {
      const v16h* bp = reinterpret_cast<const v16h*>(sW + (kt * 4 + j) * 512);
      acc[j] = __builtin_amdgcn_wmma_f32_16x16x32_f16(
          false, a, false, bp[lane], (short)0, acc[j], false, false);
    }
  }

  int rbase = r0 + ((lane >> 4) << 3);
  int c0 = lane & 15;
#pragma unroll
  for (int j = 0; j < 4; ++j) {
    int col = j * 16 + c0;
    float bd = bdown[col];
#pragma unroll
    for (int i = 0; i < 8; ++i) {
      float v = acc[j][i] + bd;
      act[(size_t)(rbase + i) * DD + col] = gelu_tanh(v);
    }
  }
}

// ---------------------------------------------------------------------------
// Kernel 2: scores[r] = max_m(act_r . nq_m) / (||act_r|| + 1e-12)
// ---------------------------------------------------------------------------
__global__ __launch_bounds__(256) void score_kernel(
    const float* __restrict__ act, const float* __restrict__ mq,
    float* __restrict__ scores) {
  __shared__ float snq[MM * DD];
  int t = threadIdx.x;
  for (int i = t; i < MM * DD; i += 256) snq[i] = mq[i];
  __syncthreads();
  if (t < MM) {
    float ss = 0.f;
    for (int c = 0; c < DD; ++c) { float v = snq[t * DD + c]; ss += v * v; }
    float inv = 1.0f / (sqrtf(ss) + 1e-12f);
    for (int c = 0; c < DD; ++c) snq[t * DD + c] *= inv;
  }
  __syncthreads();
  int r = blockIdx.x * 256 + t;
  const float* ar = act + (size_t)r * DD;
  float xv[DD];
  float ss = 0.f;
  for (int c = 0; c < DD; c += 4) {
    float4 f = *reinterpret_cast<const float4*>(ar + c);
    xv[c] = f.x; xv[c + 1] = f.y; xv[c + 2] = f.z; xv[c + 3] = f.w;
    ss += f.x * f.x + f.y * f.y + f.z * f.z + f.w * f.w;
  }
  float best = -INFINITY;
  for (int m = 0; m < MM; ++m) {
    float d = 0.f;
    for (int c = 0; c < DD; ++c) d += xv[c] * snq[m * DD + c];
    best = fmaxf(best, d);
  }
  scores[r] = best / (sqrtf(ss) + 1e-12f);
}

// ---------------------------------------------------------------------------
// Kernel 3: top-K=64 per batch (iterative argmax, lowest-index tie-break),
// then ascending sort of the selected indices.
// ---------------------------------------------------------------------------
__global__ __launch_bounds__(256) void topk_kernel(
    const float* __restrict__ scores, int* __restrict__ idx) {
  __shared__ float sc[NN];
  __shared__ float rv[256];
  __shared__ int   ri[256];
  __shared__ int   sel[KSEL];
  int b = blockIdx.x, t = threadIdx.x;
  for (int i = t; i < NN; i += 256) sc[i] = scores[(size_t)b * NN + i];
  __syncthreads();
  for (int j = 0; j < KSEL; ++j) {
    float bv = -INFINITY; int bi = NN;
    for (int i = t; i < NN; i += 256) {
      float v = sc[i];
      if (v > bv) { bv = v; bi = i; }   // strict > keeps lowest index
    }
    rv[t] = bv; ri[t] = bi;
    __syncthreads();
    for (int s = 128; s > 0; s >>= 1) {
      if (t < s) {
        if (rv[t + s] > rv[t] || (rv[t + s] == rv[t] && ri[t + s] < ri[t])) {
          rv[t] = rv[t + s]; ri[t] = ri[t + s];
        }
      }
      __syncthreads();
    }
    if (t == 0) { sel[j] = ri[0]; sc[ri[0]] = -INFINITY; }
    __syncthreads();
  }
  if (t == 0) {
    for (int i = 1; i < KSEL; ++i) {
      int v = sel[i]; int k2 = i - 1;
      while (k2 >= 0 && sel[k2] > v) { sel[k2 + 1] = sel[k2]; --k2; }
      sel[k2 + 1] = v;
    }
    for (int i2 = 0; i2 < KSEL; ++i2) idx[b * KSEL + i2] = sel[i2];
  }
}

// ---------------------------------------------------------------------------
// Kernel 4: gather + LayerNorm + 4-head MHA + out_proj; scatter results back
// into act at the selected rows. One block per batch; tiny fp32 VALU work.
// ---------------------------------------------------------------------------
__global__ __launch_bounds__(256) void attn_kernel(
    float* __restrict__ act, const int* __restrict__ idx,
    const float* __restrict__ mq,
    const float* __restrict__ lnw, const float* __restrict__ lnb,
    const float* __restrict__ ipw, const float* __restrict__ ipb,
    const float* __restrict__ opw, const float* __restrict__ opb) {
  extern __shared__ float smem[];
  float* sseq = smem;           // 80*64
  float* sln  = sseq + 5120;    // 80*64
  float* sq   = sln + 5120;     // 80*64
  float* sk   = sq + 5120;      // 80*64
  float* sv   = sk + 5120;      // 80*64
  float* sco  = sv + 5120;      // 64*80
  float* so   = sco + 5120;     // 64*64
  __shared__ int sidx[KSEL];
  int b = blockIdx.x, t = threadIdx.x;
  if (t < KSEL) sidx[t] = idx[b * KSEL + t];
  __syncthreads();

  for (int o = t; o < 80 * DD; o += 256) {
    int r = o / DD, c = o % DD;
    float v = (r < MM) ? mq[r * DD + c]
                       : act[((size_t)b * NN + sidx[r - MM]) * DD + c];
    sseq[o] = v;
  }
  __syncthreads();

  if (t < 80) {
    float mu = 0.f;
    for (int c = 0; c < DD; ++c) mu += sseq[t * DD + c];
    mu *= (1.0f / DD);
    float var = 0.f;
    for (int c = 0; c < DD; ++c) { float d = sseq[t * DD + c] - mu; var += d * d; }
    var *= (1.0f / DD);
    float inv = rsqrtf(var + 1e-5f);
    for (int c = 0; c < DD; ++c)
      sln[t * DD + c] = (sseq[t * DD + c] - mu) * inv * lnw[c] + lnb[c];
  }
  __syncthreads();

  for (int o = t; o < 80 * 192; o += 256) {   // qkv = ln @ in_w.T + in_b
    int r = o / 192, j = o % 192;
    float s = ipb[j];
    const float* wr = ipw + j * DD;
    const float* xr = sln + r * DD;
    for (int d = 0; d < DD; ++d) s += xr[d] * wr[d];
    if (j < 64)       sq[r * DD + j] = s;
    else if (j < 128) sk[r * DD + (j - 64)] = s;
    else              sv[r * DD + (j - 128)] = s;
  }
  __syncthreads();

  for (int h = 0; h < 4; ++h) {
    int hb = h * 16;
    for (int o = t; o < 64 * 80; o += 256) {  // scores for sparse query rows
      int qr = o / 80, kc = o % 80;
      float s = 0.f;
      for (int d = 0; d < 16; ++d) s += sq[(MM + qr) * DD + hb + d] * sk[kc * DD + hb + d];
      sco[o] = s * 0.25f;                     // 1/sqrt(dh=16)
    }
    __syncthreads();
    if (t < 64) {
      float mx = -INFINITY;
      for (int kc = 0; kc < 80; ++kc) mx = fmaxf(mx, sco[t * 80 + kc]);
      float sum = 0.f;
      for (int kc = 0; kc < 80; ++kc) { float e = expf(sco[t * 80 + kc] - mx); sco[t * 80 + kc] = e; sum += e; }
      float inv = 1.0f / sum;
      for (int kc = 0; kc < 80; ++kc) sco[t * 80 + kc] *= inv;
    }
    __syncthreads();
    for (int o = t; o < 64 * 16; o += 256) {  // o = attn @ v
      int qr = o / 16, d = o % 16;
      float s = 0.f;
      for (int kc = 0; kc < 80; ++kc) s += sco[qr * 80 + kc] * sv[kc * DD + hb + d];
      so[qr * DD + hb + d] = s;
    }
    __syncthreads();
  }

  for (int o = t; o < 64 * 64; o += 256) {    // out_proj + scatter into act
    int qr = o / 64, c = o % 64;
    float s = opb[c];
    const float* wr = opw + c * DD;
    const float* xr = so + qr * DD;
    for (int d = 0; d < DD; ++d) s += xr[d] * wr[d];
    act[((size_t)b * NN + sidx[qr]) * DD + c] = s;
  }
}

// ---------------------------------------------------------------------------
// Kernel 5: out = X + gamma * (act @ W_up + b_up), WMMA f16->f32.
// 4096 blocks x 8 waves; wave handles 8 nt groups (16x128 output) x 2 k-steps.
// Result staged in 64KB LDS tile so residual-add + store is coalesced b128.
// ---------------------------------------------------------------------------
__global__ __launch_bounds__(256) void up_kernel(
    const float* __restrict__ act, const float* __restrict__ X,
    const _Float16* __restrict__ fWu, const float* __restrict__ bup,
    const float* __restrict__ gammap, float* __restrict__ out) {
  extern __shared__ float sout[];   // 16*1024 floats
  int t = threadIdx.x, wave = t >> 5, lane = t & 31;
  size_t r0 = (size_t)blockIdx.x * 16;
  const float* arow = act + (r0 + (lane & 15)) * DD;
  int kofs = (lane >= 16) ? 8 : 0;

  v16h a0, a1;
  {
    float4 fa = *reinterpret_cast<const float4*>(arow + kofs);
    float4 fb = *reinterpret_cast<const float4*>(arow + kofs + 4);
    float4 fc = *reinterpret_cast<const float4*>(arow + kofs + 16);
    float4 fd = *reinterpret_cast<const float4*>(arow + kofs + 20);
    a0[0]  = (_Float16)fa.x; a0[1]  = (_Float16)fa.y; a0[2]  = (_Float16)fa.z; a0[3]  = (_Float16)fa.w;
    a0[4]  = (_Float16)fb.x; a0[5]  = (_Float16)fb.y; a0[6]  = (_Float16)fb.z; a0[7]  = (_Float16)fb.w;
    a0[8]  = (_Float16)fc.x; a0[9]  = (_Float16)fc.y; a0[10] = (_Float16)fc.z; a0[11] = (_Float16)fc.w;
    a0[12] = (_Float16)fd.x; a0[13] = (_Float16)fd.y; a0[14] = (_Float16)fd.z; a0[15] = (_Float16)fd.w;
    fa = *reinterpret_cast<const float4*>(arow + 32 + kofs);
    fb = *reinterpret_cast<const float4*>(arow + 32 + kofs + 4);
    fc = *reinterpret_cast<const float4*>(arow + 32 + kofs + 16);
    fd = *reinterpret_cast<const float4*>(arow + 32 + kofs + 20);
    a1[0]  = (_Float16)fa.x; a1[1]  = (_Float16)fa.y; a1[2]  = (_Float16)fa.z; a1[3]  = (_Float16)fa.w;
    a1[4]  = (_Float16)fb.x; a1[5]  = (_Float16)fb.y; a1[6]  = (_Float16)fb.z; a1[7]  = (_Float16)fb.w;
    a1[8]  = (_Float16)fc.x; a1[9]  = (_Float16)fc.y; a1[10] = (_Float16)fc.z; a1[11] = (_Float16)fc.w;
    a1[12] = (_Float16)fd.x; a1[13] = (_Float16)fd.y; a1[14] = (_Float16)fd.z; a1[15] = (_Float16)fd.w;
  }

  float gam = gammap[0];
  int c0 = lane & 15, rsel = (lane >> 4) << 3;
#pragma unroll
  for (int g = 0; g < 8; ++g) {
    int nt = wave * 8 + g;
    v8f acc = {};
    const v16h* b0 = reinterpret_cast<const v16h*>(fWu + (size_t)nt * 512);
    const v16h* b1 = reinterpret_cast<const v16h*>(fWu + (size_t)(64 + nt) * 512);
    acc = __builtin_amdgcn_wmma_f32_16x16x32_f16(false, a0, false, b0[lane], (short)0, acc, false, false);
    acc = __builtin_amdgcn_wmma_f32_16x16x32_f16(false, a1, false, b1[lane], (short)0, acc, false, false);
    int col = nt * 16 + c0;
    float bu = bup[col];
#pragma unroll
    for (int i = 0; i < 8; ++i)
      sout[(rsel + i) * CIN + col] = gam * (acc[i] + bu);
  }
  __syncthreads();

  const float* xb = X + r0 * CIN;
  float* ob = out + r0 * CIN;
  for (int i = t; i < 4096; i += 256) {
    int e = i * 4;
    float4 xv = *reinterpret_cast<const float4*>(xb + e);
    float4 sv4 = *reinterpret_cast<const float4*>(sout + e);
    float4 r;
    r.x = xv.x + sv4.x; r.y = xv.y + sv4.y; r.z = xv.z + sv4.z; r.w = xv.w + sv4.w;
    *reinterpret_cast<float4*>(ob + e) = r;
  }
}

// ---------------------------------------------------------------------------
extern "C" void kernel_launch(void* const* d_in, const int* in_sizes, int n_in,
                              void* d_out, int out_size, void* d_ws, size_t ws_size,
                              hipStream_t stream) {
  const float* X    = (const float*)d_in[0];
  const float* Wd   = (const float*)d_in[1];
  const float* bd   = (const float*)d_in[2];
  const float* Wu   = (const float*)d_in[3];
  const float* bu   = (const float*)d_in[4];
  const float* mq   = (const float*)d_in[5];
  const float* lnw  = (const float*)d_in[6];
  const float* lnb  = (const float*)d_in[7];
  const float* ipw  = (const float*)d_in[8];
  const float* ipb  = (const float*)d_in[9];
  const float* opw  = (const float*)d_in[10];
  const float* opb  = (const float*)d_in[11];
  const float* gam  = (const float*)d_in[12];
  float* out = (float*)d_out;

  char* ws = (char*)d_ws;
  _Float16* fWd = (_Float16*)ws;                                   // 131072 B
  _Float16* fWu = (_Float16*)(ws + 131072);                        // 131072 B
  float* act    = (float*)(ws + 262144);                           // 16 MB
  float* scores = (float*)(ws + 262144 + 16777216);                // 256 KB
  int*   idxp   = (int*)(ws + 262144 + 16777216 + 262144);         // 4 KB

  pack_weights_kernel<<<512, 256, 0, stream>>>(Wd, Wu, fWd, fWu);
  down_gelu_kernel<<<512, 256, 131072, stream>>>(X, bd, fWd, act);
  score_kernel<<<ROWS / 256, 256, 0, stream>>>(act, mq, scores);
  topk_kernel<<<BB, 256, 0, stream>>>(scores, idxp);
  attn_kernel<<<BB, 256, 139264, stream>>>(act, idxp, mq, lnw, lnb, ipw, ipb, opw, opb);
  up_kernel<<<ROWS / 16, 256, 65536, stream>>>(act, X, fWu, bu, gam, out);
}
